// FilterTokenLayer_15693810499636
// MI455X (gfx1250) — matrix operand order repaired and checked by reading, not compiled, not run on hardware
//
#include <hip/hip_runtime.h>

// Problem constants (from the reference)
#define B_      64
#define C_      256
#define N_      4096
#define T_      16
#define G_      8          // N-chunks per batch
#define NCHUNK  512        // N_ / G_

typedef __bf16 bf16_t;
typedef bf16_t v16bf __attribute__((ext_vector_type(16)));
typedef float  v8f   __attribute__((ext_vector_type(8)));
typedef int    v4i   __attribute__((ext_vector_type(4)));

union F8  { v8f   v; float        f[8]; };
union BFr { v16bf v; unsigned int u[8]; };

#define AS1 __attribute__((address_space(1)))
#define AS3 __attribute__((address_space(3)))

#if defined(__has_builtin)
#if __has_builtin(__builtin_amdgcn_global_load_async_to_lds_b128)
#define HAVE_ASYNC_LDS 1
#endif
#endif
#ifndef HAVE_ASYNC_LDS
#define HAVE_ASYNC_LDS 0
#endif

// f32 -> bf16 (round-to-nearest-even), packed pair into one dword
__device__ __forceinline__ unsigned int pack_bf16(float lo, float hi) {
  unsigned int ul = __float_as_uint(lo);
  unsigned int uh = __float_as_uint(hi);
  ul += 0x7FFFu + ((ul >> 16) & 1u);
  uh += 0x7FFFu + ((uh >> 16) & 1u);
  return (ul >> 16) | (uh & 0xFFFF0000u);
}
__device__ __forceinline__ unsigned short bf16_one(float f) {
  unsigned int u = __float_as_uint(f);
  u += 0x7FFFu + ((u >> 16) & 1u);
  return (unsigned short)(u >> 16);
}

// One block = one (batch b, N-chunk g). 256 threads = 8 wave32s.
// Phase 1: scores(512x16) = x^T * W^T via v_wmma_f32_16x16x32_bf16, track col max
// Phase 2: p = exp(s - m), column sums; store p^T (16x512) bf16 in LDS
// Phase 3: partial out (16x256) = p^T * x^T via WMMA; x restaged per 32-n K-block
//          into LDS with GLOBAL_LOAD_ASYNC_TO_LDS_B128 (ASYNCcnt) when available.
__global__ __launch_bounds__(256)
void ftl_partial(const float* __restrict__ x, const float* __restrict__ Wt,
                 float* __restrict__ accp, float* __restrict__ ms)
{
  __shared__ __align__(16) unsigned char smem[51264];
  float*          scores  = (float*)(smem);                   // [512][16] f32 (ph1/2)
  float*          xstageF = (float*)(smem);                   // [256][32] f32 (ph3, aliased)
  unsigned short* PtH     = (unsigned short*)(smem + 32768);  // [16][512] bf16
  unsigned int*   PtU     = (unsigned int*)(smem + 32768);
  float*          wavemax = (float*)(smem + 49152);           // [8][16][2]
  float*          mbuf    = (float*)(smem + 50176);           // [16]
  float*          sumbuf  = (float*)(smem + 50240);           // [16][16]

  const int tid  = threadIdx.x;
  const int lane = tid & 31;
  const int w    = tid >> 5;        // wave id 0..7
  const int b    = blockIdx.y;
  const int g    = blockIdx.x;
  const int n0g  = g * NCHUNK;

  const float* xb = x + (size_t)b * C_ * N_;

  // lane roles per the CDNA5 wave32 WMMA layouts
  const int tcol   = lane & 15;                // B/C/D column, A row
  const int kbaseA = (lane < 16) ? 0 : 8;      // table-order A K half (unused)
  const int kbaseB = (lane < 16) ? 0 : 16;     // K half for A and B (this code's contiguous k order)
  const int rbase  = (lane < 16) ? 0 : 8;      // C/D row base

  // ---- Preload GEMM1 B-operand (W) fragments: B[k][t] = W[t][c0+k], bf16 in VGPRs
  BFr wf[8];
  #pragma unroll
  for (int kb = 0; kb < 8; ++kb) {
    #pragma unroll
    for (int v = 0; v < 8; ++v) {
      const float* pw = Wt + tcol * C_ + kb * 32 + kbaseB + 2 * v;
      wf[kb].u[v] = pack_bf16(pw[0], pw[1]);
    }
  }

  // ---- Phase 1: scores for 4 n-tiles per wave -------------------------------
  float lmax = -3.402823e38f;
  #pragma unroll 1
  for (int i = 0; i < 4; ++i) {
    const int nt = w * 4 + i;
    const int n0 = n0g + nt * 16;
    F8 s = {};
    #pragma unroll 2
    for (int kb = 0; kb < 8; ++kb) {
      BFr a;                                    // A[m][k] = x[c0+k][n0+m]
      #pragma unroll
      for (int v = 0; v < 8; ++v) {
        const int k = 2 * v + kbaseB;
        const float* pa = xb + (size_t)(kb * 32 + k) * N_ + n0 + tcol;
        a.u[v] = pack_bf16(pa[0], pa[N_]);
      }
      s.v = __builtin_amdgcn_wmma_f32_16x16x32_bf16(false, a.v, false, wf[kb].v,
                                                    (short)0, s.v, false, false);
    }
    #pragma unroll
    for (int r = 0; r < 8; ++r) {
      scores[(nt * 16 + rbase + r) * 16 + tcol] = s.f[r];
      lmax = fmaxf(lmax, s.f[r]);
    }
  }
  wavemax[(w * 16 + tcol) * 2 + (lane >> 4)] = lmax;
  __syncthreads();

  if (tid < 16) {                               // per-t chunk max
    float m = -3.402823e38f;
    #pragma unroll
    for (int ww = 0; ww < 8; ++ww) {
      m = fmaxf(m, wavemax[(ww * 16 + tid) * 2 + 0]);
      m = fmaxf(m, wavemax[(ww * 16 + tid) * 2 + 1]);
    }
    mbuf[tid] = m;
  }
  __syncthreads();

  // ---- Phase 2: p = exp(s - m), column sums, p^T as bf16 --------------------
  {
    const int t2  = tid & 15;
    const int grp = tid >> 4;                   // 16 groups x 32 rows
    const float mt = mbuf[t2];
    float lsum = 0.f;
    #pragma unroll 4
    for (int i2 = 0; i2 < 32; ++i2) {
      const int n = grp * 32 + i2;
      const float e = __expf(scores[n * 16 + t2] - mt);
      lsum += e;
      PtH[t2 * 512 + n] = bf16_one(e);
    }
    sumbuf[t2 * 16 + grp] = lsum;
  }
  __syncthreads();
  if (tid < 16) {
    float ssum = 0.f;
    #pragma unroll
    for (int j = 0; j < 16; ++j) ssum += sumbuf[tid * 16 + j];
    float* msp = ms + (((size_t)b * G_ + g) * T_ + tid) * 2;
    msp[0] = mbuf[tid];
    msp[1] = ssum;
  }

  // ---- Phase 3: partial out = p^T * x^T, K = 512 in 16 blocks of 32 ---------
  F8 acc2[2] = {};
  #pragma unroll 1
  for (int kb = 0; kb < 16; ++kb) {
    __syncthreads();                            // prior readers done / phase-2 done
    {
      // stage x[:, n0g + kb*32 .. +31] (f32) into LDS, one c-row per thread
      const float* gp = xb + (size_t)tid * N_ + n0g + kb * 32;
      float*       lp = xstageF + tid * 32;
#if HAVE_ASYNC_LDS
      #pragma unroll
      for (int q = 0; q < 8; ++q) {
        __builtin_amdgcn_global_load_async_to_lds_b128(
            (AS1 v4i*)(unsigned long long)(const void*)(gp + q * 4),
            (AS3 v4i*)(unsigned)(unsigned long long)(void*)(lp + q * 4),
            0, 0);
      }
#if defined(__has_builtin) && __has_builtin(__builtin_amdgcn_s_wait_asynccnt)
      __builtin_amdgcn_s_wait_asynccnt(0);
#else
      asm volatile("s_wait_asynccnt 0" ::: "memory");
#endif
#else
      const float4* p4 = (const float4*)gp;
      #pragma unroll
      for (int q = 0; q < 8; ++q)
        ((float4*)lp)[q] = p4[q];
      if (kb + 1 < 16) __builtin_prefetch((const void*)(p4 + 8), 0, 0);
#endif
    }
    __syncthreads();

    BFr a2;                                     // A2[t][k] = p[n0g+kb*32+k][t]
    #pragma unroll
    for (int v = 0; v < 8; ++v) {
      const int k = 2 * v + kbaseB;
      a2.u[v] = PtU[tcol * 256 + kb * 16 + (k >> 1)];
    }
    #pragma unroll
    for (int j = 0; j < 2; ++j) {               // 2 c-tiles per wave
      const int c1 = (w * 2 + j) * 16;
      BFr b2;                                   // B2[k][c] = x[c][n0g+kb*32+k]
      #pragma unroll
      for (int v = 0; v < 8; ++v) {
        const float2 p2 =
            *(const float2*)&xstageF[(c1 + tcol) * 32 + kbaseB + 2 * v];
        b2.u[v] = pack_bf16(p2.x, p2.y);
      }
      acc2[j].v = __builtin_amdgcn_wmma_f32_16x16x32_bf16(false, a2.v, false, b2.v,
                                                          (short)0, acc2[j].v,
                                                          false, false);
    }
  }

  // write chunk partials (unnormalized)
  float* ap = accp + (((size_t)b * G_ + g) * T_) * C_;
  #pragma unroll
  for (int j = 0; j < 2; ++j) {
    const int cc = (w * 2 + j) * 16 + tcol;
    #pragma unroll
    for (int r = 0; r < 8; ++r)
      ap[(rbase + r) * C_ + cc] = acc2[j].f[r];
  }
}

// Merge the G_ chunk partials per (b,t,c): flash-style rescale + normalize.
__global__ __launch_bounds__(256)
void ftl_finalize(const float* __restrict__ accp, const float* __restrict__ ms,
                  float* __restrict__ out)
{
  const int idx = blockIdx.x * 256 + threadIdx.x;   // flat (b,t,c)
  const int c = idx & (C_ - 1);
  const int t = (idx >> 8) & (T_ - 1);
  const int b = idx >> 12;

  float mg[G_], sg[G_];
  float M = -3.402823e38f;
  #pragma unroll
  for (int g = 0; g < G_; ++g) {
    const float* p = ms + (((size_t)b * G_ + g) * T_ + t) * 2;
    mg[g] = p[0]; sg[g] = p[1];
    M = fmaxf(M, mg[g]);
  }
  float denom = 0.f, val = 0.f;
  #pragma unroll
  for (int g = 0; g < G_; ++g) {
    const float sc = __expf(mg[g] - M);
    denom += sc * sg[g];
    val   += sc * accp[(((size_t)b * G_ + g) * T_ + t) * C_ + c];
  }
  out[idx] = val / denom;
}

extern "C" void kernel_launch(void* const* d_in, const int* in_sizes, int n_in,
                              void* d_out, int out_size, void* d_ws, size_t ws_size,
                              hipStream_t stream) {
  (void)in_sizes; (void)n_in; (void)out_size;
  const float* x  = (const float*)d_in[0];   // (B, C, N) f32
  const float* Wt = (const float*)d_in[1];   // (T, C)    f32
  float* out  = (float*)d_out;               // (B, T, C) f32
  float* accp = (float*)d_ws;                                   // B*G*T*C f32 = 8 MB
  float* ms   = accp + (size_t)B_ * G_ * T_ * C_;               // B*G*T*2  f32
  if (ws_size < ((size_t)B_ * G_ * T_ * C_ + (size_t)B_ * G_ * T_ * 2) * sizeof(float))
    return;

  ftl_partial<<<dim3(G_, B_), 256, 0, stream>>>(x, Wt, accp, ms);
  ftl_finalize<<<(B_ * T_ * C_) / 256, 256, 0, stream>>>(accp, ms, out);
}
